// Rapidash_85667417686345
// MI455X (gfx1250) — matrix operand order repaired and checked by reading, not compiled, not run on hardware
//
#include <hip/hip_runtime.h>
#include <hip/hip_bf16.h>

// ---------------------------------------------------------------------------
// Problem constants (match reference)
// ---------------------------------------------------------------------------
#define NN      16384      // nodes
#define NB      8          // graphs
#define GG      2048       // nodes per graph
#define KK      16         // knn
#define EE      (NN*KK)    // edges = 262144
#define CC      128        // hidden
#define BDD     128        // basis dim
#define WFF     4
#define LL      4

typedef unsigned short u16;
typedef __attribute__((ext_vector_type(16))) __bf16 v16bf;
typedef __attribute__((ext_vector_type(8)))  __bf16 v8bf;
typedef __attribute__((ext_vector_type(8)))  float  f32x8;
typedef int i32x4 __attribute__((vector_size(16)));

// CDNA5 async global->LDS movers (ASYNCcnt-tracked), guarded so the file still
// compiles (with the synchronous fallback) on toolchains lacking the builtins.
#if defined(__has_builtin)
#if __has_builtin(__builtin_amdgcn_global_load_async_to_lds_b128) && \
    __has_builtin(__builtin_amdgcn_global_load_async_to_lds_b32)  && \
    __has_builtin(__builtin_amdgcn_s_wait_asynccnt)
#define USE_ASYNC_LDS 1
#endif
#endif
#ifndef USE_ASYNC_LDS
#define USE_ASYNC_LDS 0
#endif

#define GAS __attribute__((address_space(1)))
#define LAS __attribute__((address_space(3)))

// typed AS casts for the async builtins (signature: int* / int4* pointees)
#define G_I32(p)  ((GAS int*)(GAS void*)(p))
#define L_I32(p)  ((LAS int*)(LAS void*)(p))
#define G_I128(p) ((GAS i32x4*)(GAS void*)(p))
#define L_I128(p) ((LAS i32x4*)(LAS void*)(p))

// ---------------------------------------------------------------------------
// Helpers
// ---------------------------------------------------------------------------
__device__ __forceinline__ u16 f2bf(float x) {
    unsigned u = __float_as_uint(x);
    unsigned r = (u + 0x7FFFu + ((u >> 16) & 1u)) >> 16;
    return (u16)r;
}

__device__ __forceinline__ float gelu_f(float x) {
    const float c = 0.7978845608028654f;          // sqrt(2/pi)
    float t = tanhf(c * (x + 0.044715f * x * x * x));
    return 0.5f * x * (1.0f + t);
}

__device__ __forceinline__ v16bf cat16(v8bf a, v8bf b) {
    return __builtin_shufflevector(a, b, 0,1,2,3,4,5,6,7,8,9,10,11,12,13,14,15);
}

// Load one WMMA A fragment (16x32 bf16) directly from a row-major bf16 matrix.
// lane<16:  row = base+lane,     K = kt*32 + {0..7, 16..23}
// lane>=16: row = base+lane-16,  K = kt*32 + {8..15, 24..31}
__device__ __forceinline__ v16bf load_afrag(const u16* A, size_t rowBase, int K,
                                            int kt, int lane) {
    size_t r = rowBase + (lane & 15);
    const u16* p = A + r * (size_t)K + kt * 32 + ((lane & 16) ? 8 : 0);
    v8bf a = *(const v8bf*)p;
    v8bf b = *(const v8bf*)(p + 16);
    return cat16(a, b);
}

// Load one pre-packed WMMA B fragment (32x16 bf16): 16 contiguous bf16 per lane.
__device__ __forceinline__ v16bf load_bfrag(const u16* P, int kt, int NT, int nt,
                                            int lane) {
    const u16* p = P + (((size_t)kt * NT + nt) * 32 + lane) * 16;
    v8bf a = *(const v8bf*)p;
    v8bf b = *(const v8bf*)(p + 8);
    return cat16(a, b);
}

// ---------------------------------------------------------------------------
// kNN: one thread per node; graph positions staged in LDS (async movers when
// available); top-16 by d2 via register insertion sort.
// ---------------------------------------------------------------------------
__global__ __launch_bounds__(256) void knn_kernel(const float* __restrict__ pos,
                                                  int* __restrict__ srcIdx) {
    __shared__ float px[GG], py[GG], pz[GG];
    int t = threadIdx.x;
    int nodeBase = blockIdx.x * 256;
    int g = nodeBase / GG;
    int gbase = g * GG;
#if USE_ASYNC_LDS
    for (int i = t; i < GG; i += 256) {
        const float* gp = pos + (size_t)(gbase + i) * 3;
        __builtin_amdgcn_global_load_async_to_lds_b32(G_I32(gp + 0), L_I32(&px[i]), 0, 0);
        __builtin_amdgcn_global_load_async_to_lds_b32(G_I32(gp + 1), L_I32(&py[i]), 0, 0);
        __builtin_amdgcn_global_load_async_to_lds_b32(G_I32(gp + 2), L_I32(&pz[i]), 0, 0);
    }
    __builtin_amdgcn_s_wait_asynccnt(0);
#else
    for (int i = t; i < GG; i += 256) {
        px[i] = pos[(size_t)(gbase + i) * 3 + 0];
        py[i] = pos[(size_t)(gbase + i) * 3 + 1];
        pz[i] = pos[(size_t)(gbase + i) * 3 + 2];
    }
#endif
    __syncthreads();

    int n  = nodeBase + t;
    int ln = n - gbase;
    float ox = px[ln], oy = py[ln], oz = pz[ln];

    float bd[KK];
    int   bi[KK];
#pragma unroll
    for (int k = 0; k < KK; ++k) { bd[k] = 3.4e38f; bi[k] = ln; }

    for (int j = 0; j < GG; ++j) {
        float dx = px[j] - ox, dy = py[j] - oy, dz = pz[j] - oz;
        float d2 = dx * dx + dy * dy + dz * dz;
        if (d2 < bd[KK - 1]) {
            bd[KK - 1] = d2; bi[KK - 1] = j;
#pragma unroll
            for (int s = KK - 1; s > 0; --s) {
                if (bd[s] < bd[s - 1]) {
                    float td = bd[s]; bd[s] = bd[s - 1]; bd[s - 1] = td;
                    int   ti = bi[s]; bi[s] = bi[s - 1]; bi[s - 1] = ti;
                }
            }
        }
    }
#pragma unroll
    for (int k = 0; k < KK; ++k)
        srcIdx[(size_t)n * KK + k] = gbase + bi[k];
}

// ---------------------------------------------------------------------------
// Pack a row-major f32 [K,Nc] B matrix into per-lane WMMA bf16 fragments:
// P[(kt*NT+nt)*32*16 + lane*16 + j] = W[kt*32 + (lane&16?8:0) + (j&7) + (j&8?16:0)][nt*16 + lane&15]
// ---------------------------------------------------------------------------
__global__ void pack_b_kernel(const float* __restrict__ W, u16* __restrict__ P,
                              int K, int Nc) {
    int idx = blockIdx.x * 256 + threadIdx.x;
    if (idx >= K * Nc) return;
    int j    = idx & 15;
    int lane = (idx >> 4) & 31;
    int tt   = idx >> 9;
    int NT   = Nc >> 4;
    int nt   = tt % NT;
    int kt   = tt / NT;
    int k    = kt * 32 + ((lane & 16) ? 8 : 0) + (j & 7) + ((j & 8) ? 16 : 0);
    int col  = nt * 16 + (lane & 15);
    P[idx] = f2bf(W[(size_t)k * Nc + col]);
}

// ---------------------------------------------------------------------------
// h = x @ W_embed   (thread per (n,c); K=65)
// ---------------------------------------------------------------------------
__global__ void embed_kernel(const float* __restrict__ x,
                             const float* __restrict__ We,
                             float* __restrict__ h) {
    int idx = blockIdx.x * 256 + threadIdx.x;
    int c = idx & 127;
    int n = idx >> 7;
    if (n >= NN) return;
    float s = 0.f;
    for (int k = 0; k < 65; ++k)
        s += x[(size_t)n * 65 + k] * We[(size_t)k * CC + c];
    h[(size_t)n * CC + c] = s;
}

// ---------------------------------------------------------------------------
// Edge basis layer 1: kb = gelu(poly(d) @ Wb1 + bb1)  -> bf16 [E,128]
// thread per (edge, channel)
// ---------------------------------------------------------------------------
__global__ void edge_basis_kernel(const float* __restrict__ pos,
                                  const int* __restrict__ srcIdx,
                                  const float* __restrict__ Wb1,
                                  const float* __restrict__ bb1,
                                  u16* __restrict__ kb) {
    size_t idx = (size_t)blockIdx.x * 256 + threadIdx.x;
    int c = (int)(idx & 127);
    size_t e = idx >> 7;
    if (e >= (size_t)EE) return;
    int s = srcIdx[e];
    int d = (int)(e >> 4);   // dst node (edges contiguous per node)
    float dx = pos[(size_t)s * 3 + 0] - pos[(size_t)d * 3 + 0];
    float dy = pos[(size_t)s * 3 + 1] - pos[(size_t)d * 3 + 1];
    float dz = pos[(size_t)s * 3 + 2] - pos[(size_t)d * 3 + 2];
    float dd = sqrtf(dx * dx + dy * dy + dz * dz + 1e-12f);
    float d2 = dd * dd, d3 = d2 * dd;
    float v = dd * Wb1[c] + d2 * Wb1[BDD + c] + d3 * Wb1[2 * BDD + c] + bb1[c];
    kb[e * BDD + c] = f2bf(gelu_f(v));
}

// ---------------------------------------------------------------------------
// Generic WMMA GEMM:  out = epi(A[M,KD] @ B[KD,NC] + bias)
// A bf16 row-major, B pre-packed bf16 fragments, grid.y tiles NC in 128-col chunks.
// EPI 0: gelu -> bf16 store.  EPI 1: += into f32 buffer (residual).
// One wave computes a 16x128 output tile.
// ---------------------------------------------------------------------------
template <int KD, int NC, int EPI>
__global__ __launch_bounds__(256) void wmma_gemm_kernel(
        const u16* __restrict__ A, const u16* __restrict__ Bp,
        const float* __restrict__ bias, u16* __restrict__ outB,
        float* __restrict__ outF, int M) {
    int lane = threadIdx.x & 31;
    int w    = threadIdx.x >> 5;
    int rt   = blockIdx.x * 8 + w;
    size_t rowBase = (size_t)rt * 16;
    if (rowBase >= (size_t)M) return;
    int cy = blockIdx.y;
    constexpr int NT = NC / 16;

    // Prefetch the next row-tile's A stream (global_prefetch_b8).
    if (rowBase + 128 < (size_t)M)
        __builtin_prefetch(A + (rowBase + 128) * KD + (size_t)lane * KD / 2, 0, 1);

    f32x8 acc[8] = {};
#pragma unroll
    for (int kt = 0; kt < KD / 32; ++kt) {
        v16bf a = load_afrag(A, rowBase, KD, kt, lane);
#pragma unroll
        for (int t = 0; t < 8; ++t) {
            v16bf b = load_bfrag(Bp, kt, NT, cy * 8 + t, lane);
            acc[t] = __builtin_amdgcn_wmma_f32_16x16x32_bf16(
                false, a, false, b, (short)0, acc[t], false, false);
        }
    }

#pragma unroll
    for (int t = 0; t < 8; ++t) {
        int col = cy * 128 + t * 16 + (lane & 15);
        float bv = bias[col];
#pragma unroll
        for (int v = 0; v < 8; ++v) {
            size_t row = rowBase + v + ((lane & 16) ? 8 : 0);
            float xv = acc[t][v] + bv;
            if (EPI == 0) {
                outB[row * NC + col] = f2bf(gelu_f(xv));
            } else {
                outF[row * NC + col] += xv;
            }
        }
    }
}

// ---------------------------------------------------------------------------
// Fused depthwise-conv + segment-sum + LayerNorm.
// One wave per destination node: its 16 contiguous edges form one WMMA M-tile.
//   gather h[src] rows into LDS with async global->LDS movers (ASYNCcnt),
//   k_i (16 edges x 128 ch) via 4x8 v_wmma_f32_16x16x32_bf16,
//   multiply by gathered h[src] (padded LDS stride -> conflict free),
//   reduce over edges (intra-lane + shfl_xor 16), LayerNorm, bf16 z out.
// ---------------------------------------------------------------------------
#define HSTRIDE 140   // floats; 140*4 B: 16B-aligned rows, halves land 32 banks apart

__global__ __launch_bounds__(128) void conv_ln_kernel(
        const u16* __restrict__ kerB,      // [E,128] bf16
        const u16* __restrict__ WkP,       // packed Wk (this layer)
        const int* __restrict__ srcIdx,    // [E]
        const float* __restrict__ h,       // [N,128]
        const float* __restrict__ ln_g, const float* __restrict__ ln_b,
        u16* __restrict__ z) {             // [N,128] bf16
    __shared__ float hl[4][KK][HSTRIDE];
    int lane = threadIdx.x & 31;
    int wv   = threadIdx.x >> 5;
    int n    = blockIdx.x * 4 + wv;
    float (*Lrow)[HSTRIDE] = hl[wv];
    size_t eBase = (size_t)n * KK;

    // gather h rows of the 16 source neighbors into LDS
#if USE_ASYNC_LDS
    for (int r = 0; r < KK; ++r) {
        int sr = srcIdx[eBase + r];
        __builtin_amdgcn_global_load_async_to_lds_b128(
            G_I128(h + (size_t)sr * CC + lane * 4),
            L_I128(&Lrow[r][lane * 4]), 0, 0);
    }
#else
    for (int r = 0; r < KK; ++r) {
        int sr = srcIdx[eBase + r];
        float4 v = *(const float4*)(h + (size_t)sr * CC + lane * 4);
        *(float4*)(&Lrow[r][lane * 4]) = v;
    }
#endif

    // A fragments: 16 edge-rows of the spatial kernel (bf16) — overlaps with
    // the in-flight async LDS gathers.
    v16bf af[4];
#pragma unroll
    for (int kt = 0; kt < 4; ++kt)
        af[kt] = load_afrag(kerB, eBase, BDD, kt, lane);

#if USE_ASYNC_LDS
    __builtin_amdgcn_s_wait_asynccnt(0);
#endif

    int ebase = (lane & 16) ? 8 : 0;
    float myagg[8];
#pragma unroll
    for (int ct = 0; ct < 8; ++ct) {
        f32x8 acc = {};
#pragma unroll
        for (int kt = 0; kt < 4; ++kt) {
            v16bf b = load_bfrag(WkP, kt, 8, ct, lane);
            acc = __builtin_amdgcn_wmma_f32_16x16x32_bf16(
                false, af[kt], false, b, (short)0, acc, false, false);
        }
        int chan = ct * 16 + (lane & 15);
        float s = 0.f;
#pragma unroll
        for (int v = 0; v < 8; ++v)
            s += acc[v] * Lrow[ebase + v][chan];
        s += __shfl_xor(s, 16, 32);          // combine edge halves
        myagg[ct] = s;                        // lanes l & l+16 now duplicate
    }

    // LayerNorm over 128 channels (8 per lane across 16 lanes, halves dup)
    float sum = 0.f, ssq = 0.f;
#pragma unroll
    for (int ct = 0; ct < 8; ++ct) { sum += myagg[ct]; ssq += myagg[ct] * myagg[ct]; }
    for (int o = 8; o >= 1; o >>= 1) {
        sum += __shfl_xor(sum, o, 32);
        ssq += __shfl_xor(ssq, o, 32);
    }
    float mean = sum * (1.0f / CC);
    float var  = ssq * (1.0f / CC) - mean * mean;
    float rstd = rsqrtf(var + 1e-5f);

    if (lane < 16) {
#pragma unroll
        for (int ct = 0; ct < 8; ++ct) {
            int chan = ct * 16 + lane;
            float zv = (myagg[ct] - mean) * rstd * ln_g[chan] + ln_b[chan];
            z[(size_t)n * CC + chan] = f2bf(zv);
        }
    }
}

// ---------------------------------------------------------------------------
// readout: one wave per node, dot(h[n], Wr) + br
// ---------------------------------------------------------------------------
__global__ void readout_kernel(const float* __restrict__ h,
                               const float* __restrict__ Wr,
                               const float* __restrict__ br,
                               float* __restrict__ out) {
    int lane = threadIdx.x & 31;
    int wid  = (blockIdx.x * blockDim.x + threadIdx.x) >> 5;
    if (wid >= NN) return;
    float s = 0.f;
#pragma unroll
    for (int j = 0; j < 4; ++j)
        s += h[(size_t)wid * CC + lane * 4 + j] * Wr[lane * 4 + j];
    for (int o = 16; o >= 1; o >>= 1) s += __shfl_xor(s, o, 32);
    if (lane == 0) out[wid] = s + br[0];
}

// ---------------------------------------------------------------------------
// Host launcher
// ---------------------------------------------------------------------------
extern "C" void kernel_launch(void* const* d_in, const int* in_sizes, int n_in,
                              void* d_out, int out_size, void* d_ws, size_t ws_size,
                              hipStream_t stream) {
    const float* x       = (const float*)d_in[0];
    const float* pos     = (const float*)d_in[1];
    const float* W_embed = (const float*)d_in[3];
    const float* Wb1     = (const float*)d_in[4];
    const float* bb1     = (const float*)d_in[5];
    const float* Wb2     = (const float*)d_in[6];
    const float* bb2     = (const float*)d_in[7];
    const float* Wk      = (const float*)d_in[8];
    const float* ln_g    = (const float*)d_in[9];
    const float* ln_b    = (const float*)d_in[10];
    const float* W1      = (const float*)d_in[11];
    const float* b1      = (const float*)d_in[12];
    const float* W2      = (const float*)d_in[13];
    const float* b2      = (const float*)d_in[14];
    const float* Wr      = (const float*)d_in[15];
    const float* br      = (const float*)d_in[16];
    float* out = (float*)d_out;

    // workspace carve-up
    char* ws = (char*)d_ws;
    size_t off = 0;
    auto alloc = [&](size_t bytes) -> char* {
        char* p = ws + off;
        off += (bytes + 255) & ~(size_t)255;
        return p;
    };
    int*   srcIdx = (int*)  alloc((size_t)EE * 4);
    u16*   kerB   = (u16*)  alloc((size_t)EE * BDD * 2);   // 64 MB spatial kernel
    u16*   kbu    = (u16*)  alloc((size_t)EE * BDD * 2);   // kb, later reused for u
    float* h      = (float*)alloc((size_t)NN * CC * 4);
    u16*   z      = (u16*)  alloc((size_t)NN * CC * 2);
    u16*   Wb2P   = (u16*)  alloc((size_t)BDD * BDD * 2);
    u16*   WkP    = (u16*)  alloc((size_t)LL * BDD * CC * 2);
    u16*   W1P    = (u16*)  alloc((size_t)LL * CC * (WFF * CC) * 2);
    u16*   W2P    = (u16*)  alloc((size_t)LL * (WFF * CC) * CC * 2);
    u16*   u      = kbu;    // [N,512] bf16 reuses kb space after GEMM-E2 consumes kb
    (void)ws_size; (void)in_sizes; (void)n_in; (void)out_size;

    // 1) kNN graph
    knn_kernel<<<NN / 256, 256, 0, stream>>>(pos, srcIdx);

    // 2) pack all B operands to WMMA fragment layout (bf16)
    pack_b_kernel<<<(BDD * BDD) / 256, 256, 0, stream>>>(Wb2, Wb2P, BDD, BDD);
    for (int l = 0; l < LL; ++l) {
        pack_b_kernel<<<(BDD * CC) / 256, 256, 0, stream>>>(
            Wk + (size_t)l * BDD * CC, WkP + (size_t)l * BDD * CC, BDD, CC);
        pack_b_kernel<<<(CC * WFF * CC) / 256, 256, 0, stream>>>(
            W1 + (size_t)l * CC * WFF * CC, W1P + (size_t)l * CC * WFF * CC, CC, WFF * CC);
        pack_b_kernel<<<(WFF * CC * CC) / 256, 256, 0, stream>>>(
            W2 + (size_t)l * WFF * CC * CC, W2P + (size_t)l * WFF * CC * CC, WFF * CC, CC);
    }

    // 3) h = x @ W_embed
    embed_kernel<<<(NN * CC) / 256, 256, 0, stream>>>(x, W_embed, h);

    // 4) edge basis: kb = gelu(poly @ Wb1 + bb1)
    edge_basis_kernel<<<(int)(((size_t)EE * BDD) / 256), 256, 0, stream>>>(
        pos, srcIdx, Wb1, bb1, kbu);

    // 5) kernel = gelu(kb @ Wb2 + bb2)   [WMMA]
    wmma_gemm_kernel<BDD, CC, 0><<<dim3(EE / 128, 1), 256, 0, stream>>>(
        kbu, Wb2P, bb2, kerB, nullptr, EE);

    // 6) L ConvNext layers
    for (int l = 0; l < LL; ++l) {
        conv_ln_kernel<<<NN / 4, 128, 0, stream>>>(
            kerB, WkP + (size_t)l * BDD * CC, srcIdx, h,
            ln_g + (size_t)l * CC, ln_b + (size_t)l * CC, z);
        // u = gelu(z @ W1 + b1)   [WMMA], 512 cols in 4 chunks
        wmma_gemm_kernel<CC, WFF * CC, 0><<<dim3(NN / 128, WFF), 256, 0, stream>>>(
            z, W1P + (size_t)l * CC * WFF * CC, b1 + (size_t)l * WFF * CC,
            u, nullptr, NN);
        // h += u @ W2 + b2        [WMMA residual]
        wmma_gemm_kernel<WFF * CC, CC, 1><<<dim3(NN / 128, 1), 256, 0, stream>>>(
            u, W2P + (size_t)l * WFF * CC * CC, b2 + (size_t)l * CC,
            nullptr, h, NN);
    }

    // 7) readout
    readout_kernel<<<NN / 8, 256, 0, stream>>>(h, Wr, br, out);
}
